// Model_34660386079360
// MI455X (gfx1250) — compile-verified, matrix operand order checked by reading, not run on hardware
//
#include <hip/hip_runtime.h>
#include <hip/hip_bf16.h>

typedef __attribute__((ext_vector_type(16))) _Float16 v16h;
typedef __attribute__((ext_vector_type(8)))  _Float16 v8h;
typedef __attribute__((ext_vector_type(8)))  float    v8f;

#define GRID_H 653
#define GRID_W 574
#define N_NODES 10000
#define N_EDGES 160000
#define D_EMB 256
#define N_HEADS 4
#define EDGE_CHUNK 20000

// ---------------------------------------------------------------------------
// float -> monotonic uint key (for atomicMax-based segment max)
// ---------------------------------------------------------------------------
__device__ __forceinline__ unsigned fkey(float f) {
    unsigned u = __float_as_uint(f);
    return (u & 0x80000000u) ? ~u : (u | 0x80000000u);
}
__device__ __forceinline__ float funkey(unsigned k) {
    unsigned u = (k & 0x80000000u) ? (k & 0x7FFFFFFFu) : ~k;
    return __uint_as_float(u);
}

// ---------------------------------------------------------------------------
// WMMA helpers.
// CDNA5 16-bit A/B fragment: lane m=L&15 selects the row (A) / column (Bt);
// lanes 0-15 hold K {0..7,16..23}, lanes 16-31 hold K {8..15,24..31}
// (hs = L>>4 selects the +8 offset) -> two aligned 16B loads per fragment.
// ---------------------------------------------------------------------------
__device__ __forceinline__ v16h load_frag(const _Float16* __restrict__ p) {
    v8h lo = *(const v8h*)(p);
    v8h hi = *(const v8h*)(p + 16);
    v16h r;
#pragma unroll
    for (int e = 0; e < 8; ++e) { r[e] = lo[e]; r[8 + e] = hi[e]; }
    return r;
}
__device__ __forceinline__ v8f wmma_f16(v16h a, v16h b, v8f c) {
    return __builtin_amdgcn_wmma_f32_16x16x32_f16(false, a, false, b,
                                                  (short)0, c, false, false);
}

// ---------------------------------------------------------------------------
// Fused per-node ROI CNN: ROI gather -> (conv3x3 relu + maxpool2 'SAME') x3
// Pool windows (stride 2) are disjoint so each conv output is computed once.
// Output: flattened 3*3*128 = 1152 f16 features per node.
// ---------------------------------------------------------------------------
__global__ __launch_bounds__(128) void cnn_conv_kernel(
    const float* __restrict__ grid, const int* __restrict__ pos,
    const float* __restrict__ W1, const float* __restrict__ b1,
    const float* __restrict__ W2, const float* __restrict__ b2,
    const float* __restrict__ W3, const float* __restrict__ b3,
    _Float16* __restrict__ feat)
{
    __shared__ float s_roi[17 * 17 * 2];
    __shared__ float s_c1[9 * 9 * 32];
    __shared__ float s_c2[5 * 5 * 64];
    const int node = blockIdx.x;
    const int tid  = threadIdx.x;
    const int p0 = pos[node * 2 + 0];
    const int p1 = pos[node * 2 + 1];

    for (int i = tid; i < 17 * 17 * 2; i += 128) {
        int c = i & 1, xy = i >> 1, x = xy % 17, y = xy / 17;
        int gy = p0 + y - 8, gx = p1 + x - 8;
        float v = 0.f;
        if (gy >= 0 && gy < GRID_H && gx >= 0 && gx < GRID_W)
            v = grid[((size_t)gy * GRID_W + gx) * 2 + c];
        s_roi[(y * 17 + x) * 2 + c] = v;
    }
    __syncthreads();

    // conv1 (2->32) + pool -> 9x9x32
    for (int i = tid; i < 9 * 9 * 32; i += 128) {
        int oc = i & 31, cell = i >> 5, px = cell % 9, py = cell / 9;
        float m = -1e30f;
        for (int dy = 0; dy < 2; ++dy) {
            int y = py * 2 + dy; if (y >= 17) break;
            for (int dx = 0; dx < 2; ++dx) {
                int x = px * 2 + dx; if (x >= 17) break;
                float acc = b1[oc];
                for (int ky = 0; ky < 3; ++ky) {
                    int iy = y + ky - 1; if ((unsigned)iy >= 17u) continue;
                    for (int kx = 0; kx < 3; ++kx) {
                        int ix = x + kx - 1; if ((unsigned)ix >= 17u) continue;
                        const float* wp = W1 + ((ky * 3 + kx) * 2) * 32 + oc;
                        acc += s_roi[(iy * 17 + ix) * 2 + 0] * wp[0];
                        acc += s_roi[(iy * 17 + ix) * 2 + 1] * wp[32];
                    }
                }
                m = fmaxf(m, fmaxf(acc, 0.f));
            }
        }
        s_c1[(py * 9 + px) * 32 + oc] = m;
    }
    __syncthreads();

    // conv2 (32->64) + pool -> 5x5x64
    for (int i = tid; i < 5 * 5 * 64; i += 128) {
        int oc = i & 63, cell = i >> 6, px = cell % 5, py = cell / 5;
        float m = -1e30f;
        for (int dy = 0; dy < 2; ++dy) {
            int y = py * 2 + dy; if (y >= 9) break;
            for (int dx = 0; dx < 2; ++dx) {
                int x = px * 2 + dx; if (x >= 9) break;
                float acc = b2[oc];
                for (int ky = 0; ky < 3; ++ky) {
                    int iy = y + ky - 1; if ((unsigned)iy >= 9u) continue;
                    for (int kx = 0; kx < 3; ++kx) {
                        int ix = x + kx - 1; if ((unsigned)ix >= 9u) continue;
                        const float* ip = &s_c1[(iy * 9 + ix) * 32];
                        const float* wp = W2 + ((ky * 3 + kx) * 32) * 64 + oc;
#pragma unroll 8
                        for (int ic = 0; ic < 32; ++ic) acc += ip[ic] * wp[ic * 64];
                    }
                }
                m = fmaxf(m, fmaxf(acc, 0.f));
            }
        }
        s_c2[(py * 5 + px) * 64 + oc] = m;
    }
    __syncthreads();

    // conv3 (64->128) + pool -> 3x3x128, store f16 flattened
    for (int i = tid; i < 3 * 3 * 128; i += 128) {
        int oc = i & 127, cell = i >> 7, px = cell % 3, py = cell / 3;
        float m = -1e30f;
        for (int dy = 0; dy < 2; ++dy) {
            int y = py * 2 + dy; if (y >= 5) break;
            for (int dx = 0; dx < 2; ++dx) {
                int x = px * 2 + dx; if (x >= 5) break;
                float acc = b3[oc];
                for (int ky = 0; ky < 3; ++ky) {
                    int iy = y + ky - 1; if ((unsigned)iy >= 5u) continue;
                    for (int kx = 0; kx < 3; ++kx) {
                        int ix = x + kx - 1; if ((unsigned)ix >= 5u) continue;
                        const float* ip = &s_c2[(iy * 5 + ix) * 64];
                        const float* wp = W3 + ((ky * 3 + kx) * 64) * 128 + oc;
#pragma unroll 8
                        for (int ic = 0; ic < 64; ++ic) acc += ip[ic] * wp[ic * 128];
                    }
                }
                m = fmaxf(m, fmaxf(acc, 0.f));
            }
        }
        feat[(size_t)node * 1152 + (py * 3 + px) * 128 + oc] = (_Float16)m;
    }
}

// ---------------------------------------------------------------------------
// f32 -> f16 weight convert + transpose:  Wt[n*K + k] = W[k*N + n]
// ---------------------------------------------------------------------------
__global__ void convert_transpose_f16(const float* __restrict__ W,
                                      _Float16* __restrict__ Wt, int K, int N)
{
    int idx = blockIdx.x * 256 + threadIdx.x;
    if (idx < K * N) {
        int k = idx / N, n = idx % N;
        Wt[(size_t)n * K + k] = (_Float16)W[idx];
    }
}

// ---------------------------------------------------------------------------
// Generic WMMA GEMM:  C[M,N] = act(A[M,K] @ B[K,N] + bias), Bt = B transposed.
// Wave tile 32x32 (4 accumulators, 4 WMMAs per K-step, A/B frags each reused
// twice). Block = 256 threads = 8 waves arranged 4(M) x 2(N): 128x64 tile.
// GATHER mode: A row e is concat(emb[src[e]], emb[dst[e]]), K == 512; the
// src/dst halves are separate loops so no per-iteration selects remain.
// All branch conditions are wave-uniform -> EXEC stays all-1s around WMMA.
// ---------------------------------------------------------------------------
template<bool GATHER>
__global__ __launch_bounds__(256) void wmma_gemm_kernel(
    const _Float16* __restrict__ A,
    const _Float16* __restrict__ gatherBase,
    const int* __restrict__ gsrc, const int* __restrict__ gdst, int gRowOff,
    const _Float16* __restrict__ Bt,
    const float* __restrict__ bias,
    _Float16* __restrict__ Ch, float* __restrict__ Cf,
    int M, int N, int K, int relu,
    long long strideA, long long strideB, long long strideC)
{
    const int z = blockIdx.z;
    if (!GATHER) A += (size_t)z * strideA;
    Bt += (size_t)z * strideB;

    const int wave = threadIdx.x >> 5;
    const int lane = threadIdx.x & 31;
    const int wy = wave >> 1, wx = wave & 1;
    const int row0 = blockIdx.x * 128 + wy * 32;
    const int col0 = blockIdx.y * 64 + wx * 32;
    if (row0 >= M || col0 >= N) return;

    const int m  = lane & 15;
    const int hs = lane >> 4;
    const int r0 = min(row0 + m, M - 1);        // clamped: safe loads on edge tile
    const int r1 = min(row0 + 16 + m, M - 1);

    const _Float16* b0 = Bt + (size_t)(col0 + m) * K + hs * 8;
    const _Float16* b1 = Bt + (size_t)(col0 + 16 + m) * K + hs * 8;

    v8f c00 = {0.f,0.f,0.f,0.f,0.f,0.f,0.f,0.f};
    v8f c01 = c00, c10 = c00, c11 = c00;

    if (!GATHER) {
        const _Float16* a0 = A + (size_t)r0 * K + hs * 8;
        const _Float16* a1 = A + (size_t)r1 * K + hs * 8;
#pragma unroll 2
        for (int k0 = 0; k0 < K; k0 += 32) {
            v16h A0 = load_frag(a0 + k0);
            v16h A1 = load_frag(a1 + k0);
            v16h B0 = load_frag(b0 + k0);
            v16h B1 = load_frag(b1 + k0);
            c00 = wmma_f16(A0, B0, c00);
            c01 = wmma_f16(A0, B1, c01);
            c10 = wmma_f16(A1, B0, c10);
            c11 = wmma_f16(A1, B1, c11);
        }
    } else {
        const int e0 = gRowOff + r0, e1 = gRowOff + r1;
        const _Float16* a0s = gatherBase + (size_t)gsrc[e0] * D_EMB + hs * 8;
        const _Float16* a1s = gatherBase + (size_t)gsrc[e1] * D_EMB + hs * 8;
        const _Float16* a0d = gatherBase + (size_t)gdst[e0] * D_EMB + hs * 8;
        const _Float16* a1d = gatherBase + (size_t)gdst[e1] * D_EMB + hs * 8;
#pragma unroll 2
        for (int k0 = 0; k0 < D_EMB; k0 += 32) {        // src half: k 0..255
            v16h A0 = load_frag(a0s + k0);
            v16h A1 = load_frag(a1s + k0);
            v16h B0 = load_frag(b0 + k0);
            v16h B1 = load_frag(b1 + k0);
            c00 = wmma_f16(A0, B0, c00);
            c01 = wmma_f16(A0, B1, c01);
            c10 = wmma_f16(A1, B0, c10);
            c11 = wmma_f16(A1, B1, c11);
        }
#pragma unroll 2
        for (int k0 = 0; k0 < D_EMB; k0 += 32) {        // dst half: k 256..511
            v16h A0 = load_frag(a0d + k0);
            v16h A1 = load_frag(a1d + k0);
            v16h B0 = load_frag(b0 + D_EMB + k0);
            v16h B1 = load_frag(b1 + D_EMB + k0);
            c00 = wmma_f16(A0, B0, c00);
            c01 = wmma_f16(A0, B1, c01);
            c10 = wmma_f16(A1, B0, c10);
            c11 = wmma_f16(A1, B1, c11);
        }
    }

    // Epilogue: C/D layout -> lane n=L&15, rows r + 8*(L>>4)
    const int n  = lane & 15;
    const int mh = hs * 8;
    const float bv0 = bias ? bias[col0 + n] : 0.f;
    const float bv1 = bias ? bias[col0 + 16 + n] : 0.f;
    const bool row1ok = (row0 + 16) < M;   // M % 16 == 0 -> sub-tile granular
#pragma unroll
    for (int r = 0; r < 8; ++r) {
        size_t o0 = (size_t)z * strideC + (size_t)(row0 + mh + r) * N + col0 + n;
        float v00 = c00[r] + bv0, v01 = c01[r] + bv1;
        if (relu) { v00 = fmaxf(v00, 0.f); v01 = fmaxf(v01, 0.f); }
        if (Cf) { Cf[o0] = v00; Cf[o0 + 16] = v01; }
        if (Ch) { Ch[o0] = (_Float16)v00; Ch[o0 + 16] = (_Float16)v01; }
        if (row1ok) {
            size_t o1 = o0 + (size_t)16 * N;
            float v10 = c10[r] + bv0, v11 = c11[r] + bv1;
            if (relu) { v10 = fmaxf(v10, 0.f); v11 = fmaxf(v11, 0.f); }
            if (Cf) { Cf[o1] = v10; Cf[o1 + 16] = v11; }
            if (Ch) { Ch[o1] = (_Float16)v10; Ch[o1 + 16] = (_Float16)v11; }
        }
    }
}

// ---------------------------------------------------------------------------
// GAT attention pieces
// ---------------------------------------------------------------------------
__global__ void escore_kernel(const _Float16* __restrict__ h,   // [4][N][256]
                              const float* __restrict__ a_src,
                              const float* __restrict__ a_dst,
                              float* __restrict__ es, float* __restrict__ ed, int N)
{
    int idx = blockIdx.x * 256 + threadIdx.x;
    if (idx >= N_HEADS * N) return;
    int hd = idx / N, n = idx % N;
    const _Float16* hp = h + ((size_t)hd * N + n) * D_EMB;
    const float* as = a_src + hd * D_EMB;
    const float* ad = a_dst + hd * D_EMB;
    float s1 = 0.f, s2 = 0.f;
    for (int d = 0; d < D_EMB; ++d) {
        float v = (float)hp[d];
        s1 += v * as[d];
        s2 += v * ad[d];
    }
    es[idx] = s1;
    ed[idx] = s2;
}

__global__ void fill_u32_kernel(unsigned* __restrict__ p, unsigned v, long long n)
{
    long long i = (long long)blockIdx.x * 256 + threadIdx.x;
    if (i < n) p[i] = v;
}

__global__ void edge_max_kernel(const int* __restrict__ src, const int* __restrict__ dst,
                                const float* __restrict__ es, const float* __restrict__ ed,
                                float* __restrict__ ebuf, unsigned* __restrict__ segmax,
                                int E, int N)
{
    int idx = blockIdx.x * 256 + threadIdx.x;
    if (idx >= N_HEADS * E) return;
    int hd = idx / E, e = idx % E;
    float v = es[hd * N + src[e]] + ed[hd * N + dst[e]];
    v = (v > 0.f) ? v : 0.2f * v;          // leaky_relu 0.2
    ebuf[idx] = v;
    atomicMax(&segmax[hd * N + dst[e]], fkey(v));
}

__global__ void edge_expsum_kernel(const int* __restrict__ dst,
                                   float* __restrict__ ebuf,
                                   const unsigned* __restrict__ segmax,
                                   float* __restrict__ segsum, int E, int N)
{
    int idx = blockIdx.x * 256 + threadIdx.x;
    if (idx >= N_HEADS * E) return;
    int hd = idx / E, e = idx % E;
    float mx = funkey(segmax[hd * N + dst[e]]);
    float w = __expf(ebuf[idx] - mx);
    ebuf[idx] = w;
    atomicAdd(&segsum[hd * N + dst[e]], w);
}

// one block per edge; 256 threads = 4 heads x 64 four-wide chunks
__global__ __launch_bounds__(256) void edge_agg_kernel(
    const int* __restrict__ src, const int* __restrict__ dst,
    const float* __restrict__ ebuf, const float* __restrict__ segsum,
    const _Float16* __restrict__ h, float* __restrict__ cat, int E, int N)
{
    int e = blockIdx.x, t = threadIdx.x;
    int hd = t >> 6, d0 = (t & 63) * 4;
    int s = src[e], dn = dst[e];
    float alpha = ebuf[(size_t)hd * E + e] / segsum[hd * N + dn];
    const _Float16* hp = h + ((size_t)hd * N + s) * D_EMB + d0;
    float* cp = cat + (size_t)dn * (N_HEADS * D_EMB) + hd * D_EMB + d0;
#pragma unroll
    for (int j = 0; j < 4; ++j)
        atomicAdd(&cp[j], (float)hp[j] * alpha);
}

__global__ void relu_cvt_kernel(const float* __restrict__ in,
                                _Float16* __restrict__ out, long long n)
{
    long long i = (long long)blockIdx.x * 256 + threadIdx.x;
    if (i < n) out[i] = (_Float16)fmaxf(in[i], 0.f);
}

// final edge-MLP layer: out[e] = dot(h2[e,:], w3) + b3
__global__ void edge_out_kernel(const _Float16* __restrict__ h2,
                                const float* __restrict__ w3,
                                const float* __restrict__ b3,
                                float* __restrict__ out, int M)
{
    int i = blockIdx.x * 256 + threadIdx.x;
    if (i >= M) return;
    const _Float16* p = h2 + (size_t)i * D_EMB;
    float s = b3[0];
    for (int d = 0; d < D_EMB; ++d) s += (float)p[d] * w3[d];
    out[i] = s;
}

// ---------------------------------------------------------------------------
// Host driver
// ---------------------------------------------------------------------------
extern "C" void kernel_launch(void* const* d_in, const int* in_sizes, int n_in,
                              void* d_out, int out_size, void* d_ws, size_t ws_size,
                              hipStream_t stream)
{
    (void)in_sizes; (void)n_in; (void)out_size; (void)ws_size;

    // ---- inputs (setup_inputs() dict insertion order, leaves flattened) ----
    const float* grid = (const float*)d_in[0];
    const int*   pos  = (const int*)d_in[1];
    const int*   src  = (const int*)d_in[2];
    const int*   dst  = (const int*)d_in[3];
    const float* c1w = (const float*)d_in[4];  const float* c1b = (const float*)d_in[5];
    const float* c2w = (const float*)d_in[6];  const float* c2b = (const float*)d_in[7];
    const float* c3w = (const float*)d_in[8];  const float* c3b = (const float*)d_in[9];
    const float* dW  = (const float*)d_in[10]; const float* dB  = (const float*)d_in[11];
    // gat layer l: base 12 + l*9 -> Wn, a_src, a_dst, (W,b)x3
    const float* eW0 = (const float*)d_in[30]; const float* eB0 = (const float*)d_in[31];
    const float* eW1 = (const float*)d_in[32]; const float* eB1 = (const float*)d_in[33];
    const float* eW2 = (const float*)d_in[34]; const float* eB2 = (const float*)d_in[35];

    // ---- workspace layout ----
    char* base = (char*)d_ws;
    size_t off = 0;
    auto alloc = [&](size_t bytes) -> char* {
        char* r = base + off;
        off += (bytes + 255) & ~(size_t)255;
        return r;
    };
    _Float16* wdT   = (_Float16*)alloc(1152 * 256 * 2);
    _Float16* wnT[2], *g0T[2], *g1T[2], *g2T[2];
    for (int l = 0; l < 2; ++l) {
        wnT[l] = (_Float16*)alloc((size_t)N_HEADS * 256 * 256 * 2);
        g0T[l] = (_Float16*)alloc((size_t)1024 * 512 * 2);
        g1T[l] = (_Float16*)alloc((size_t)512 * 256 * 2);
        g2T[l] = (_Float16*)alloc((size_t)256 * 256 * 2);
    }
    _Float16* e0T   = (_Float16*)alloc((size_t)512 * 512 * 2);
    _Float16* e1T   = (_Float16*)alloc((size_t)512 * 256 * 2);
    _Float16* feat  = (_Float16*)alloc((size_t)N_NODES * 1152 * 2);
    _Float16* embA  = (_Float16*)alloc((size_t)N_NODES * 256 * 2);
    _Float16* embB  = (_Float16*)alloc((size_t)N_NODES * 256 * 2);
    _Float16* hbuf  = (_Float16*)alloc((size_t)N_HEADS * N_NODES * 256 * 2);
    float*    es    = (float*)alloc((size_t)N_HEADS * N_NODES * 4);
    float*    ed    = (float*)alloc((size_t)N_HEADS * N_NODES * 4);
    float*    ebuf  = (float*)alloc((size_t)N_HEADS * N_EDGES * 4);
    unsigned* segmx = (unsigned*)alloc((size_t)N_HEADS * N_NODES * 4);
    float*    segsm = (float*)alloc((size_t)N_HEADS * N_NODES * 4);
    float*    catf  = (float*)alloc((size_t)N_NODES * 1024 * 4);
    _Float16* cath  = (_Float16*)alloc((size_t)N_NODES * 1024 * 2);
    _Float16* m1    = (_Float16*)alloc((size_t)N_NODES * 512 * 2);
    _Float16* m2    = (_Float16*)alloc((size_t)N_NODES * 256 * 2);
    _Float16* eh1   = (_Float16*)alloc((size_t)EDGE_CHUNK * 512 * 2);
    _Float16* eh2   = (_Float16*)alloc((size_t)EDGE_CHUNK * 256 * 2);

    auto cvtT = [&](const float* W, _Float16* Wt, int K, int N) {
        int cnt = K * N;
        convert_transpose_f16<<<(cnt + 255) / 256, 256, 0, stream>>>(W, Wt, K, N);
    };
    auto gemm = [&](const _Float16* A, const _Float16* Bt, const float* bias,
                    _Float16* Ch, float* Cf, int M, int N, int K, int relu,
                    int batch, long long sA, long long sB, long long sC) {
        dim3 g((M + 127) / 128, N / 64, batch);
        wmma_gemm_kernel<false><<<g, 256, 0, stream>>>(
            A, nullptr, nullptr, nullptr, 0, Bt, bias, Ch, Cf,
            M, N, K, relu, sA, sB, sC);
    };
    auto gemm_gather = [&](const _Float16* emb, const int* gs, const int* gd,
                           int growoff, const _Float16* Bt, const float* bias,
                           _Float16* Ch, int M, int N, int K, int relu) {
        dim3 g((M + 127) / 128, N / 64, 1);
        wmma_gemm_kernel<true><<<g, 256, 0, stream>>>(
            nullptr, emb, gs, gd, growoff, Bt, bias, Ch, nullptr,
            M, N, K, relu, 0, 0, 0);
    };
    auto fill = [&](void* p, unsigned v, long long n) {
        fill_u32_kernel<<<(unsigned)((n + 255) / 256), 256, 0, stream>>>((unsigned*)p, v, n);
    };

    // ---- weight conversion (deterministic; runs every call) ----
    cvtT(dW, wdT, 1152, 256);
    for (int l = 0; l < 2; ++l) {
        const float* Wn = (const float*)d_in[12 + l * 9];
        for (int h = 0; h < N_HEADS; ++h)
            cvtT(Wn + (size_t)h * 256 * 256, wnT[l] + (size_t)h * 256 * 256, 256, 256);
        cvtT((const float*)d_in[15 + l * 9], g0T[l], 1024, 512);
        cvtT((const float*)d_in[17 + l * 9], g1T[l], 512, 256);
        cvtT((const float*)d_in[19 + l * 9], g2T[l], 256, 256);
    }
    cvtT(eW0, e0T, 512, 512);
    cvtT(eW1, e1T, 512, 256);

    // ---- stage 1: ROI CNN + dense ----
    cnn_conv_kernel<<<N_NODES, 128, 0, stream>>>(grid, pos, c1w, c1b, c2w, c2b, c3w, c3b, feat);
    gemm(feat, wdT, dB, embA, nullptr, N_NODES, 256, 1152, 1, 1, 0, 0, 0);

    // ---- stage 2: GAT layers ----
    _Float16* embIn = embA;
    _Float16* embOut = embB;
    for (int l = 0; l < 2; ++l) {
        // per-head GEMM h = emb @ Wn[h]  (batched over heads)
        gemm(embIn, wnT[l], nullptr, hbuf, nullptr,
             N_NODES, 256, 256, 0, N_HEADS, 0, 256 * 256, (long long)N_NODES * 256);
        escore_kernel<<<(N_HEADS * N_NODES + 255) / 256, 256, 0, stream>>>(
            hbuf, (const float*)d_in[13 + l * 9], (const float*)d_in[14 + l * 9],
            es, ed, N_NODES);
        fill(segmx, 0u, (long long)N_HEADS * N_NODES);
        fill(segsm, 0u, (long long)N_HEADS * N_NODES);
        fill(catf, 0u, (long long)N_NODES * 1024);
        edge_max_kernel<<<(N_HEADS * N_EDGES + 255) / 256, 256, 0, stream>>>(
            src, dst, es, ed, ebuf, segmx, N_EDGES, N_NODES);
        edge_expsum_kernel<<<(N_HEADS * N_EDGES + 255) / 256, 256, 0, stream>>>(
            dst, ebuf, segmx, segsm, N_EDGES, N_NODES);
        edge_agg_kernel<<<N_EDGES, 256, 0, stream>>>(
            src, dst, ebuf, segsm, hbuf, catf, N_EDGES, N_NODES);
        relu_cvt_kernel<<<(N_NODES * 1024 + 255) / 256, 256, 0, stream>>>(
            catf, cath, (long long)N_NODES * 1024);
        // node MLP 1024 -> 512 -> 256 -> 256
        gemm(cath, g0T[l], (const float*)d_in[16 + l * 9], m1, nullptr,
             N_NODES, 512, 1024, 1, 1, 0, 0, 0);
        gemm(m1, g1T[l], (const float*)d_in[18 + l * 9], m2, nullptr,
             N_NODES, 256, 512, 1, 1, 0, 0, 0);
        gemm(m2, g2T[l], (const float*)d_in[20 + l * 9], embOut, nullptr,
             N_NODES, 256, 256, 0, 1, 0, 0, 0);
        _Float16* t = embIn; embIn = embOut; embOut = t;
    }

    // ---- stage 3: edge MLP, chunked, input gathered on the fly ----
    for (int c0 = 0; c0 < N_EDGES; c0 += EDGE_CHUNK) {
        gemm_gather(embIn, src, dst, c0, e0T, eB0, eh1,
                    EDGE_CHUNK, 512, 512, 1);
        gemm(eh1, e1T, eB1, eh2, nullptr, EDGE_CHUNK, 256, 512, 1, 1, 0, 0, 0);
        edge_out_kernel<<<(EDGE_CHUNK + 255) / 256, 256, 0, stream>>>(
            eh2, eW2, eB2, (float*)d_out + c0, EDGE_CHUNK);
    }
}
